// GAT_19104014532842
// MI455X (gfx1250) — matrix operand order repaired and checked by reading, not compiled, not run on hardware
//
#include <hip/hip_runtime.h>
#include <hip/hip_bf16.h>
#include <math.h>
#include <stddef.h>

// GAT on MI455X (gfx1250): bf16 WMMA (16x16x32, f32 accum) for all matmuls,
// TDM (tensor_load_to_lds) async tiles for adj/e2 in the attention kernels.
#define GAT_ALPHA 0.2f
#define GAT_NEG   (-9.0e15f)

typedef __bf16 bf16_t;
typedef __attribute__((ext_vector_type(16))) __bf16 v16bf;
typedef __attribute__((ext_vector_type(8)))  float  v8f;
typedef __attribute__((ext_vector_type(4)))  unsigned int v4u;
typedef __attribute__((ext_vector_type(8)))  int v8i;
typedef __attribute__((ext_vector_type(4)))  int v4i;

static __device__ __forceinline__ float gat_leaky(float t) {
  return t > 0.f ? t : GAT_ALPHA * t;
}
static __device__ __forceinline__ float gat_elu(float t) {
  return t > 0.f ? t : (__expf(t) - 1.f);
}

// ---------------------------------------------------------------------------
// Tensor Data Mover: build D# groups (ISA 08_async_tensor Sec 8.3/8.4) and
// issue TENSOR_LOAD_TO_LDS. 2D tile: tile_d0 x tile_d1 elements, row stride
// stride0 (elements), optional LDS padding of pad rows.
static __device__ __forceinline__ void gat_tdm_load(
    unsigned lds_off, const void* gptr,
    unsigned tensor_d0, unsigned tensor_d1,
    unsigned tile_d0, unsigned tile_d1,
    unsigned long long stride0, unsigned data_size_code,
    unsigned pad_enable, unsigned pad_interval, unsigned pad_amount) {
  unsigned long long ga = (unsigned long long)(uintptr_t)gptr;
  v4u g0;
  g0[0] = 1u;  // count=1 (valid user descriptor)
  g0[1] = lds_off;
  g0[2] = (unsigned)(ga & 0xFFFFFFFFu);
  g0[3] = (unsigned)((ga >> 32) & 0x1FFFFFFu) | (2u << 30);  // type=2 ("image")
  v8i g1;
  g1[0] = (int)((data_size_code << 16) | (pad_enable << 20) |
                (pad_interval << 22) | (pad_amount << 25));
  g1[1] = (int)((tensor_d0 & 0xFFFFu) << 16);  // atomic_barrier=0 | td0[15:0]
  g1[2] = (int)(((tensor_d0 >> 16) & 0xFFFFu) | ((tensor_d1 & 0xFFFFu) << 16));
  g1[3] = (int)(((tensor_d1 >> 16) & 0xFFFFu) | ((tile_d0 & 0xFFFFu) << 16));
  g1[4] = (int)(tile_d1 & 0xFFFFu);            // tile_dim2 = 0
  g1[5] = (int)(unsigned)(stride0 & 0xFFFFFFFFu);
  g1[6] = (int)(unsigned)((stride0 >> 32) & 0xFFFFu);  // tensor_dim1_stride=0
  g1[7] = 0;
  v4i z4 = {0, 0, 0, 0};
  v8i z8 = {0, 0, 0, 0, 0, 0, 0, 0};
  // 6-arg toolchain variant: (g0, g1, g2, g3, g4, cpol)
  __builtin_amdgcn_tensor_load_to_lds(g0, g1, z4, z4, z8, 0);
}

// LDS layout for the attention kernels (single __shared__ struct -> offset 0).
#define ADJ_ROWS 16
#define ADJ_STRIDE 33  // 32 dwords + 1 pad dword (TDM pad_interval=32,amount=1)
#define ADJ_TILE_DW (ADJ_ROWS * ADJ_STRIDE)
#define LDS_E2_SIZE (4096 * 4)
#define LDS_ADJ_OFF(b) (LDS_E2_SIZE + (b) * ADJ_TILE_DW * 4)
struct AttnSmem {
  float e2s[4096];
  int adjt[2][ADJ_TILE_DW];
};
static_assert(offsetof(AttnSmem, adjt) == LDS_E2_SIZE, "LDS layout mismatch");

// ---------------------------------------------------------------------------
// Pack f32 (H, Kdim, Odim) -> bf16 B-fragment layout [h][kb][ot][lane][16].
__global__ void gat_pack_B(const float* __restrict__ src, bf16_t* __restrict__ dst,
                           int KB, int OT, int Kdim, int Odim, int total) {
  int idx = blockIdx.x * blockDim.x + threadIdx.x;
  if (idx >= total) return;
  int kk   = idx & 15;
  int lane = (idx >> 4) & 31;
  int r    = idx >> 9;
  int ot   = r % OT; r /= OT;
  int kb   = r % KB;
  int h    = r / KB;
  int K = kb * 32 + ((kk < 8) ? (((lane >> 4) << 3) + kk) : (8 + ((lane >> 4) << 3) + kk));
  int o = ot * 16 + (lane & 15);
  dst[idx] = (bf16_t)src[((size_t)h * Kdim + K) * Odim + o];
}

// Pack f32 (Nrows, Kdim) -> bf16 A-fragment layout [ib][kb][lane][16].
__global__ void gat_pack_A(const float* __restrict__ src, bf16_t* __restrict__ dst,
                           int KB, int Kdim, int total) {
  int idx = blockIdx.x * blockDim.x + threadIdx.x;
  if (idx >= total) return;
  int kk   = idx & 15;
  int lane = (idx >> 4) & 31;
  int r    = idx >> 9;
  int kb   = r % KB;
  int ib   = r / KB;
  int half = lane >> 4;
  int koff = (kk < 8) ? (half * 8 + kk) : (8 + half * 8 + kk);
  int K = kb * 32 + koff;
  int i = ib * 16 + (lane & 15);
  dst[idx] = (bf16_t)src[(size_t)i * Kdim + K];
}

// ---------------------------------------------------------------------------
// Wh[h] = x @ W[h] (4096x512 @ 512x64); register double-buffered fragments.
__global__ void gat_gemm_xW(const bf16_t* __restrict__ XA, const bf16_t* __restrict__ Wb,
                            float* __restrict__ Wh, int Nn) {
  int ib = blockIdx.x;
  int h  = blockIdx.y;
  int lane = threadIdx.x;
  v8f zero = {0.f,0.f,0.f,0.f,0.f,0.f,0.f,0.f};
  v8f acc[4] = {zero, zero, zero, zero};
  const bf16_t* abase = XA + ((size_t)ib * 16 * 32 + lane) * 16;
  const bf16_t* bbase = Wb + ((size_t)h * 16 * 4 * 32 + lane) * 16;
  v16bf a  = *(const v16bf*)(abase);
  v16bf b0 = *(const v16bf*)(bbase);
  v16bf b1 = *(const v16bf*)(bbase + 32 * 16);
  v16bf b2 = *(const v16bf*)(bbase + 2 * 32 * 16);
  v16bf b3 = *(const v16bf*)(bbase + 3 * 32 * 16);
  for (int kb = 0; kb < 16; ++kb) {
    v16bf an = a, n0 = b0, n1 = b1, n2 = b2, n3 = b3;
    if (kb < 15) {  // prefetch next-k fragments while WMMAs execute
      an = *(const v16bf*)(abase + (size_t)(kb + 1) * 32 * 16);
      const bf16_t* bn = bbase + (size_t)(kb + 1) * 4 * 32 * 16;
      n0 = *(const v16bf*)(bn);
      n1 = *(const v16bf*)(bn + 32 * 16);
      n2 = *(const v16bf*)(bn + 2 * 32 * 16);
      n3 = *(const v16bf*)(bn + 3 * 32 * 16);
    }
    acc[0] = __builtin_amdgcn_wmma_f32_16x16x32_bf16(false, a, false, b0, (short)0, acc[0], false, false);
    acc[1] = __builtin_amdgcn_wmma_f32_16x16x32_bf16(false, a, false, b1, (short)0, acc[1], false, false);
    acc[2] = __builtin_amdgcn_wmma_f32_16x16x32_bf16(false, a, false, b2, (short)0, acc[2], false, false);
    acc[3] = __builtin_amdgcn_wmma_f32_16x16x32_bf16(false, a, false, b3, (short)0, acc[3], false, false);
    a = an; b0 = n0; b1 = n1; b2 = n2; b3 = n3;
  }
  int half = lane >> 4, col = lane & 15;
#pragma unroll
  for (int ot = 0; ot < 4; ++ot)
#pragma unroll
    for (int v = 0; v < 8; ++v) {
      int row = ib * 16 + v + 8 * half;
      Wh[((size_t)h * Nn + row) * 64 + ot * 16 + col] = acc[ot][v];
    }
}

// e1/e2 row dots: one thread per (h, i). Works for both layers.
__global__ void gat_compute_e(const float* __restrict__ Wh, const float* __restrict__ a,
                              float* __restrict__ e1, float* __restrict__ e2,
                              int Nn, int Odim) {
  int idx = blockIdx.x * blockDim.x + threadIdx.x;
  int h = idx / Nn;
  const float* wh = Wh + (size_t)idx * Odim;
  const float* a1 = a + (size_t)h * 2 * Odim;
  const float* a2 = a1 + Odim;
  float s1 = 0.f, s2 = 0.f;
  for (int o = 0; o < Odim; ++o) {
    float w = wh[o];
    s1 += w * a1[o];
    s2 += w * a2[o];
  }
  e1[idx] = s1;
  e2[idx] = s2;
}

// Row softmax stats: m_i = lrelu(e1_i + max_{adj} e2_j) (leaky_relu monotonic),
// sinv_i = 1 / sum exp(e - m). One wave per (i, h).
__global__ void gat_row_stats(const int* __restrict__ adj, const float* __restrict__ e1,
                              const float* __restrict__ e2, float* __restrict__ mrow,
                              float* __restrict__ sinv, int Nn) {
  int i = blockIdx.x;
  int h = blockIdx.y;
  int lane = threadIdx.x;
  const int* arow = adj + (size_t)i * Nn;
  const float* e2h = e2 + (size_t)h * Nn;
  float e1i = e1[(size_t)h * Nn + i];
  float mx = -3.4e38f;
  for (int j = lane; j < Nn; j += 32)
    if (arow[j] > 0) mx = fmaxf(mx, e2h[j]);
#pragma unroll
  for (int d = 16; d >= 1; d >>= 1) mx = fmaxf(mx, __shfl_xor(mx, d, 32));
  float m = (mx > -3.0e38f) ? gat_leaky(e1i + mx) : GAT_NEG;
  float s = 0.f;
  for (int j = lane; j < Nn; j += 32)
    if (arow[j] > 0) s += __expf(gat_leaky(e1i + e2h[j]) - m);
#pragma unroll
  for (int d = 16; d >= 1; d >>= 1) s += __shfl_xor(s, d, 32);
  if (lane == 0) {
    mrow[(size_t)h * Nn + i] = m;
    sinv[(size_t)h * Nn + i] = (s > 0.f) ? (1.f / s) : 0.f;
  }
}

// Build the probability A-fragment from LDS-resident adj tile + e2 row.
static __device__ __forceinline__ v16bf gat_prob_frag_lds(
    const int* adjt, const float* e2s,
    float e1i, float mi, float si, int jb, int half, int rowInTile) {
  v16bf af;
  int j0 = jb * 32 + half * 8;
  int c0 = half * 8;
  const int* arow = adjt + rowInTile * ADJ_STRIDE;
#pragma unroll
  for (int q = 0; q < 8; ++q) {
    float t = gat_leaky(e1i + e2s[j0 + q]);
    float p = (arow[c0 + q] > 0) ? __expf(t - mi) * si : 0.f;
    af[q] = (bf16_t)p;
  }
#pragma unroll
  for (int q = 0; q < 8; ++q) {
    float t = gat_leaky(e1i + e2s[j0 + 16 + q]);
    float p = (arow[c0 + 16 + q] > 0) ? __expf(t - mi) * si : 0.f;
    af[8 + q] = (bf16_t)p;
  }
  return af;
}

// Layer-1 attention: helu[i][h*64+o] = elu((softmax(e) @ Wh[h])[i][o]).
// One wave per (ib, h). adj tiles + e2 row arrive via TDM into LDS;
// B fragments are register double-buffered from L2.
__global__ void gat_attn1(const int* __restrict__ adj, const float* __restrict__ e1,
                          const float* __restrict__ e2, const float* __restrict__ mrow,
                          const float* __restrict__ sinv, const bf16_t* __restrict__ WhB,
                          float* __restrict__ helu, int Nn) {
  __shared__ AttnSmem sm;
  int ib = blockIdx.x;
  int h  = blockIdx.y;
  int lane = threadIdx.x;
  int half = lane >> 4;
  int row  = lane & 15;
  int i0 = ib * 16;
  int i  = i0 + row;
  const float* e2h = e2 + (size_t)h * Nn;
  float e1i = e1[(size_t)h * Nn + i];
  float mi  = mrow[(size_t)h * Nn + i];
  float si  = sinv[(size_t)h * Nn + i];
  int JB = Nn / 32;
  // Async TDM: whole e2 row (1D) + first two adj tiles (2D, padded rows).
  gat_tdm_load(0, e2h, Nn, 1, Nn, 1, Nn, 2, 0, 0, 0);
  gat_tdm_load(LDS_ADJ_OFF(0), adj + (size_t)i0 * Nn, Nn, Nn, 32, 16, Nn, 2, 1, 4, 0);
  gat_tdm_load(LDS_ADJ_OFF(1), adj + (size_t)i0 * Nn + 32, Nn, Nn, 32, 16, Nn, 2, 1, 4, 0);

  const bf16_t* bbase = WhB + ((size_t)h * JB * 4 * 32 + lane) * 16;
  v16bf b0 = *(const v16bf*)(bbase);
  v16bf b1 = *(const v16bf*)(bbase + 32 * 16);
  v16bf b2 = *(const v16bf*)(bbase + 2 * 32 * 16);
  v16bf b3 = *(const v16bf*)(bbase + 3 * 32 * 16);
  v8f zero = {0.f,0.f,0.f,0.f,0.f,0.f,0.f,0.f};
  v8f acc[4] = {zero, zero, zero, zero};
  for (int jb = 0; jb < JB; ++jb) {
    v16bf n0 = b0, n1 = b1, n2 = b2, n3 = b3;
    if (jb + 1 < JB) {
      const bf16_t* bn = bbase + (size_t)(jb + 1) * 4 * 32 * 16;
      n0 = *(const v16bf*)(bn);
      n1 = *(const v16bf*)(bn + 32 * 16);
      n2 = *(const v16bf*)(bn + 2 * 32 * 16);
      n3 = *(const v16bf*)(bn + 3 * 32 * 16);
    }
    __builtin_amdgcn_s_wait_tensorcnt((short)1);  // oldest tile resident
    int buf = jb & 1;
    v16bf af = gat_prob_frag_lds(sm.adjt[buf], sm.e2s, e1i, mi, si, jb, half, row);
    asm volatile("s_wait_dscnt 0x0" ::: "memory");  // LDS reads done before refill
    if (jb + 2 < JB)
      gat_tdm_load(LDS_ADJ_OFF(buf), adj + (size_t)i0 * Nn + (size_t)(jb + 2) * 32,
                   Nn, Nn, 32, 16, Nn, 2, 1, 4, 0);
    acc[0] = __builtin_amdgcn_wmma_f32_16x16x32_bf16(false, af, false, b0, (short)0, acc[0], false, false);
    acc[1] = __builtin_amdgcn_wmma_f32_16x16x32_bf16(false, af, false, b1, (short)0, acc[1], false, false);
    acc[2] = __builtin_amdgcn_wmma_f32_16x16x32_bf16(false, af, false, b2, (short)0, acc[2], false, false);
    acc[3] = __builtin_amdgcn_wmma_f32_16x16x32_bf16(false, af, false, b3, (short)0, acc[3], false, false);
    b0 = n0; b1 = n1; b2 = n2; b3 = n3;
  }
  __builtin_amdgcn_s_wait_tensorcnt((short)0);
  int col = lane & 15;
#pragma unroll
  for (int ot = 0; ot < 4; ++ot)
#pragma unroll
    for (int v = 0; v < 8; ++v) {
      int r = i0 + v + 8 * half;
      helu[(size_t)r * 512 + h * 64 + ot * 16 + col] = gat_elu(acc[ot][v]);
    }
}

// Who = helu @ W_out (4096x512 @ 512x16).
__global__ void gat_gemm_hWo(const bf16_t* __restrict__ hA, const bf16_t* __restrict__ WoB,
                             float* __restrict__ Who) {
  int ib = blockIdx.x;
  int lane = threadIdx.x;
  v8f acc = {0.f,0.f,0.f,0.f,0.f,0.f,0.f,0.f};
  const bf16_t* abase = hA + ((size_t)ib * 16 * 32 + lane) * 16;
  const bf16_t* bbase = WoB + (size_t)lane * 16;
  v16bf a = *(const v16bf*)(abase);
  v16bf b = *(const v16bf*)(bbase);
  for (int kb = 0; kb < 16; ++kb) {
    v16bf an = a, bn = b;
    if (kb < 15) {
      an = *(const v16bf*)(abase + (size_t)(kb + 1) * 32 * 16);
      bn = *(const v16bf*)(bbase + (size_t)(kb + 1) * 32 * 16);
    }
    acc = __builtin_amdgcn_wmma_f32_16x16x32_bf16(false, a, false, b, (short)0, acc, false, false);
    a = an; b = bn;
  }
  int half = lane >> 4, col = lane & 15;
#pragma unroll
  for (int v = 0; v < 8; ++v)
    Who[((size_t)ib * 16 + v + 8 * half) * 16 + col] = acc[v];
}

// Layer-2 attention + ELU + log_softmax over C=16 classes.
__global__ void gat_attn2(const int* __restrict__ adj, const float* __restrict__ e1o,
                          const float* __restrict__ e2o, const float* __restrict__ mo,
                          const float* __restrict__ so, const bf16_t* __restrict__ WhoB,
                          float* __restrict__ out, int Nn) {
  __shared__ AttnSmem sm;
  int ib = blockIdx.x;
  int lane = threadIdx.x;
  int half = lane >> 4;
  int row  = lane & 15;
  int i0 = ib * 16;
  int i  = i0 + row;
  float e1i = e1o[i], mi = mo[i], si = so[i];
  int JB = Nn / 32;
  gat_tdm_load(0, e2o, Nn, 1, Nn, 1, Nn, 2, 0, 0, 0);
  gat_tdm_load(LDS_ADJ_OFF(0), adj + (size_t)i0 * Nn, Nn, Nn, 32, 16, Nn, 2, 1, 4, 0);
  gat_tdm_load(LDS_ADJ_OFF(1), adj + (size_t)i0 * Nn + 32, Nn, Nn, 32, 16, Nn, 2, 1, 4, 0);

  const bf16_t* bbase = WhoB + (size_t)lane * 16;
  v16bf b = *(const v16bf*)(bbase);
  v8f acc = {0.f,0.f,0.f,0.f,0.f,0.f,0.f,0.f};
  for (int jb = 0; jb < JB; ++jb) {
    v16bf bn = b;
    if (jb + 1 < JB)
      bn = *(const v16bf*)(bbase + (size_t)(jb + 1) * 32 * 16);
    __builtin_amdgcn_s_wait_tensorcnt((short)1);
    int buf = jb & 1;
    v16bf af = gat_prob_frag_lds(sm.adjt[buf], sm.e2s, e1i, mi, si, jb, half, row);
    asm volatile("s_wait_dscnt 0x0" ::: "memory");
    if (jb + 2 < JB)
      gat_tdm_load(LDS_ADJ_OFF(buf), adj + (size_t)i0 * Nn + (size_t)(jb + 2) * 32,
                   Nn, Nn, 32, 16, Nn, 2, 1, 4, 0);
    acc = __builtin_amdgcn_wmma_f32_16x16x32_bf16(false, af, false, b, (short)0, acc, false, false);
    b = bn;
  }
  __builtin_amdgcn_s_wait_tensorcnt((short)0);
  // ELU + log_softmax per output row (one row per 16-lane half, per VGPR).
#pragma unroll
  for (int v = 0; v < 8; ++v) {
    float val = gat_elu(acc[v]);
    float mx = val;
#pragma unroll
    for (int d = 1; d < 16; d <<= 1) mx = fmaxf(mx, __shfl_xor(mx, d, 32));
    float s = __expf(val - mx);
#pragma unroll
    for (int d = 1; d < 16; d <<= 1) s += __shfl_xor(s, d, 32);
    int r = i0 + v + 8 * half;
    out[(size_t)r * 16 + (lane & 15)] = val - mx - __logf(s);
  }
}

// ---------------------------------------------------------------------------
extern "C" void kernel_launch(void* const* d_in, const int* in_sizes, int n_in,
                              void* d_out, int out_size, void* d_ws, size_t ws_size,
                              hipStream_t stream) {
  const float* x    = (const float*)d_in[0];
  const int*   adj  = (const int*)d_in[1];
  const float* W    = (const float*)d_in[2];
  const float* a    = (const float*)d_in[3];
  const float* Wout = (const float*)d_in[4];
  const float* aout = (const float*)d_in[5];
  float* out = (float*)d_out;

  const int Nn = 4096, F = 512, O = 64, H = 8, C = 16;
  const int IB = Nn / 16, JB = Nn / 32;

  char* p = (char*)d_ws;
  auto alloc = [&](size_t bytes) {
    char* r = p;
    p += (bytes + 255) & ~(size_t)255;
    return r;
  };
  float*  Wh   = (float*)alloc((size_t)H * Nn * O * 4);
  bf16_t* Wb   = (bf16_t*)alloc((size_t)H * F * O * 2);
  bf16_t* xA   = (bf16_t*)alloc((size_t)Nn * F * 2);
  bf16_t* WhB  = (bf16_t*)alloc((size_t)H * Nn * O * 2);
  float*  e1   = (float*)alloc((size_t)H * Nn * 4);
  float*  e2   = (float*)alloc((size_t)H * Nn * 4);
  float*  mrow = (float*)alloc((size_t)H * Nn * 4);
  float*  sinv = (float*)alloc((size_t)H * Nn * 4);
  float*  helu = (float*)alloc((size_t)Nn * H * O * 4);
  bf16_t* hA   = (bf16_t*)alloc((size_t)Nn * 512 * 2);
  bf16_t* WoB  = (bf16_t*)alloc((size_t)512 * C * 2);
  float*  Who  = (float*)alloc((size_t)Nn * C * 4);
  bf16_t* WhoB = (bf16_t*)alloc((size_t)Nn * C * 2);
  float*  e1o  = (float*)alloc((size_t)Nn * 4);
  float*  e2o  = (float*)alloc((size_t)Nn * 4);
  float*  mo   = (float*)alloc((size_t)Nn * 4);
  float*  so   = (float*)alloc((size_t)Nn * 4);

  // Layer 1
  {
    int total = H * F * O;
    gat_pack_B<<<(total + 255) / 256, 256, 0, stream>>>(W, Wb, F / 32, O / 16, F, O, total);
  }
  {
    int total = Nn * F;
    gat_pack_A<<<(total + 255) / 256, 256, 0, stream>>>(x, xA, F / 32, F, total);
  }
  gat_gemm_xW<<<dim3(IB, H), 32, 0, stream>>>(xA, Wb, Wh, Nn);
  {
    int total = H * Nn * O;
    gat_pack_B<<<(total + 255) / 256, 256, 0, stream>>>(Wh, WhB, JB, O / 16, Nn, O, total);
  }
  gat_compute_e<<<(H * Nn) / 256, 256, 0, stream>>>(Wh, a, e1, e2, Nn, O);
  gat_row_stats<<<dim3(Nn, H), 32, 0, stream>>>(adj, e1, e2, mrow, sinv, Nn);
  gat_attn1<<<dim3(IB, H), 32, 0, stream>>>(adj, e1, e2, mrow, sinv, WhB, helu, Nn);

  // Layer 2
  {
    int total = Nn * 512;
    gat_pack_A<<<(total + 255) / 256, 256, 0, stream>>>(helu, hA, 512 / 32, 512, total);
  }
  {
    int total = 512 * C;
    gat_pack_B<<<(total + 255) / 256, 256, 0, stream>>>(Wout, WoB, 512 / 32, 1, 512, C, total);
  }
  gat_gemm_hWo<<<IB, 32, 0, stream>>>(hA, WoB, Who);
  gat_compute_e<<<Nn / 256, 256, 0, stream>>>(Who, aout, e1o, e2o, Nn, C);
  gat_row_stats<<<dim3(Nn, 1), 32, 0, stream>>>(adj, e1o, e2o, mo, so, Nn);
  {
    int total = Nn * C;
    gat_pack_B<<<(total + 255) / 256, 256, 0, stream>>>(Who, WhoB, JB, 1, Nn, C, total);
  }
  gat_attn2<<<IB, 32, 0, stream>>>(adj, e1o, e2o, mo, so, WhoB, out, Nn);
}
